// GroupedExperts_33268816675067
// MI455X (gfx1250) — matrix operand order repaired and checked by reading, not compiled, not run on hardware
//
#include <hip/hip_runtime.h>

// Problem constants (match reference setup_inputs()).
#define T_TOK 16384
#define NEXP  8
#define DIN   2048
#define DHID  1408

typedef __attribute__((ext_vector_type(16))) __bf16 v16bf;
typedef __attribute__((ext_vector_type(8)))  float  v8f;
typedef int v4i __attribute__((__vector_size__(16)));
typedef unsigned short u16;
typedef unsigned int   u32;

#define AS1 __attribute__((address_space(1)))
#define AS3 __attribute__((address_space(3)))

#if __has_builtin(__builtin_amdgcn_global_load_async_to_lds_b128) && \
    __has_builtin(__builtin_amdgcn_s_wait_asynccnt)
#define USE_ASYNC_LDS 1
#else
#define USE_ASYNC_LDS 0
#endif

#if __has_builtin(__builtin_amdgcn_sched_barrier)
#define SBAR() __builtin_amdgcn_sched_barrier(0)
#else
#define SBAR()
#endif

// LDS tile geometry: 128 rows x 32 elements, padded to 40-element rows (80 B)
// so b128 fragment reads hit distinct bank groups (start bank = r*20 mod 64).
constexpr int LROW  = 40;
constexpr int LTILE = 128 * LROW;

// ---------------- fp32 -> bf16 (round-to-nearest-even) ----------------
__device__ __forceinline__ u16 f2bf(float f) {
  u32 u = __float_as_uint(f);
  u32 r = u + 0x7FFFu + ((u >> 16) & 1u);
  return (u16)(r >> 16);
}

union FragBF {
  v16bf v;
  uint4 q[2];
};

__device__ __forceinline__ v8f wmma_bf16(const FragBF& a, const FragBF& b, v8f c) {
  return __builtin_amdgcn_wmma_f32_16x16x32_bf16(false, a.v, false, b.v,
                                                 (short)0, c, false, false);
}

// ---------------- x: fp32 -> bf16, same layout ----------------
__global__ void cvt_bf16_kernel(const float* __restrict__ in,
                                u16* __restrict__ out, int n) {
  int i = (blockIdx.x * blockDim.x + threadIdx.x) * 4;
  if (i >= n) return;
  float4 f = *(const float4*)(in + i);
  u32 lo = (u32)f2bf(f.x) | ((u32)f2bf(f.y) << 16);
  u32 hi = (u32)f2bf(f.z) | ((u32)f2bf(f.w) << 16);
  *(uint2*)(out + i) = make_uint2(lo, hi);
}

// ---------- weights: fp32 (K x N) -> bf16 transposed (N x K), per expert ----------
__global__ void transpose_bf16_kernel(const float* __restrict__ in,
                                      u16* __restrict__ out, int K, int N) {
  __shared__ u16 tile[32][33];
  const int e = blockIdx.z;
  in  += (size_t)e * K * N;
  out += (size_t)e * N * K;
  const int n0 = blockIdx.x * 32, k0 = blockIdx.y * 32;
  const int tx = threadIdx.x, ty = threadIdx.y;   // block = (32, 8)
#pragma unroll
  for (int i = 0; i < 32; i += 8)
    tile[ty + i][tx] = f2bf(in[(size_t)(k0 + ty + i) * N + (n0 + tx)]);
  __syncthreads();
#pragma unroll
  for (int i = 0; i < 32; i += 8)
    out[(size_t)(n0 + ty + i) * K + (k0 + tx)] = tile[tx][ty + i];
}

// ---------------- staging helpers ----------------
// Per-thread chunks: c0 = (row tid>>2, col (tid&3)*8), c1 = same col, row+64.
#if USE_ASYNC_LDS
__device__ __forceinline__ void stage_async_tile(const u16* __restrict__ g, int kk, int K,
                                                 u16* lds, int c0r, int c1r, int c0c) {
  const u16* gp0 = g + (size_t)c0r * K + kk + c0c;
  const u16* gp1 = g + (size_t)c1r * K + kk + c0c;
  __builtin_amdgcn_global_load_async_to_lds_b128(
      (AS1 v4i*)gp0, (AS3 v4i*)(lds + c0r * LROW + c0c), 0, 0);
  __builtin_amdgcn_global_load_async_to_lds_b128(
      (AS1 v4i*)gp1, (AS3 v4i*)(lds + c1r * LROW + c0c), 0, 0);
}
#else
__device__ __forceinline__ void ld2(const u16* __restrict__ g, int kk, int K,
                                    int c0r, int c1r, int c0c,
                                    uint4& q0, uint4& q1) {
  q0 = *(const uint4*)(g + (size_t)c0r * K + kk + c0c);
  q1 = *(const uint4*)(g + (size_t)c1r * K + kk + c0c);
}
__device__ __forceinline__ void st2(u16* __restrict__ lds, int c0r, int c1r, int c0c,
                                    uint4 q0, uint4 q1) {
  *(uint4*)(lds + c0r * LROW + c0c) = q0;
  *(uint4*)(lds + c1r * LROW + c0c) = q1;
}
#endif

// B fragment: lane r holds column (nt*16+r); K at hi*8..+7 and 16+hi*8..+7.
__device__ __forceinline__ void load_bfrag(FragBF& b, const u16* tilebase, int nt, int r, int hi) {
  const u16* p = tilebase + ((nt * 16 + r) * LROW + hi * 8);
  b.q[0] = *(const uint4*)(p);
  b.q[1] = *(const uint4*)(p + 16);
}

// ---------------- grouped GEMM (BM=128, BN=128, BK=32), 8 waves ----------------
// Wave tiling: 4 m-waves x 2 n-waves; each wave computes 32 rows x 64 cols,
// so each B fragment feeds two WMMAs (A row-fragment reuse).
// FUSED:  D1 = X@W1, D3 = X@W3, writes bf16 silu(D1)*D3
// !FUSED: writes fp32 X@W1
template <bool FUSED>
__global__ __launch_bounds__(256)
void grouped_gemm_kernel(const u16* __restrict__ A,     // [T x K] bf16 row-major
                         const u16* __restrict__ B1T,   // [E x N x K] bf16 (transposed)
                         const u16* __restrict__ B3T,   // [E x N x K] bf16 (FUSED only)
                         u16*   __restrict__ Hout,      // [T x N] bf16 (FUSED)
                         float* __restrict__ Fout,      // [T x N] f32  (!FUSED)
                         const int* __restrict__ m_sizes,
                         int K, int N) {
  constexpr int BM = 128, BN = 128, BK = 32;
#if USE_ASYNC_LDS
  constexpr int NBUF = 3;   // 3-deep async pipeline
#else
  constexpr int NBUF = 2;
#endif
  __shared__ __align__(16) u16 As [NBUF][LTILE];
  __shared__ __align__(16) u16 Bs1[NBUF][LTILE];
  __shared__ __align__(16) u16 Bs3[FUSED ? NBUF : 1][FUSED ? LTILE : 16];

  const int tid  = threadIdx.x;
  const int wave = tid >> 5;
  const int lane = tid & 31;
  const int r    = lane & 15;   // row/col within 16x16 tile
  const int hi   = lane >> 4;   // half-wave select
  const int mw   = wave >> 1;   // 0..3 : 32-row slice
  const int nw   = wave & 1;    // 0..1 : 64-col slice
  const int m0   = blockIdx.y * BM;
  const int n0   = blockIdx.x * BN;

  // Expert of this row tile (tiles never straddle experts: counts are mult. of 512).
  int e = 0;
  {
    int csum = m_sizes[0];
    while (m0 >= csum && e < NEXP - 1) { ++e; csum += m_sizes[e]; }
  }

  const u16* Ab  = A + (size_t)m0 * K;
  const u16* B1b = B1T + (size_t)e * N * K + (size_t)n0 * K;
  const u16* B3b = FUSED ? (B3T + (size_t)e * N * K + (size_t)n0 * K) : (const u16*)nullptr;

  const int c0r = tid >> 2;
  const int c1r = c0r + 64;
  const int c0c = (tid & 3) * 8;

  const v8f vzero = {0.f, 0.f, 0.f, 0.f, 0.f, 0.f, 0.f, 0.f};
  v8f acc1[8];                 // [mi*4 + n]
  v8f acc3[FUSED ? 8 : 1];
#pragma unroll
  for (int n = 0; n < 8; ++n) {
    acc1[n] = vzero;
    if constexpr (FUSED) acc3[n] = vzero;
  }

  const int NT = K / BK;

  // Bulk-load all fragments for a k-step, fence, then run all WMMAs.
  auto compute = [&](int cur) {
    FragBF a0, a1, b1f[4], b3f[FUSED ? 4 : 1];
    const u16* ap = &As[cur][(mw * 32 + r) * LROW + hi * 8];
    a0.q[0] = *(const uint4*)(ap);
    a0.q[1] = *(const uint4*)(ap + 16);
    a1.q[0] = *(const uint4*)(ap + 16 * LROW);
    a1.q[1] = *(const uint4*)(ap + 16 * LROW + 16);
#pragma unroll
    for (int n = 0; n < 4; ++n) {
      load_bfrag(b1f[n], Bs1[cur], nw * 4 + n, r, hi);
      if constexpr (FUSED) load_bfrag(b3f[n], Bs3[cur], nw * 4 + n, r, hi);
    }
    SBAR();   // loads above, WMMAs below: forces overlap-friendly liveness
#pragma unroll
    for (int n = 0; n < 4; ++n) {
      acc1[n]     = wmma_bf16(a0, b1f[n], acc1[n]);
      acc1[4 + n] = wmma_bf16(a1, b1f[n], acc1[4 + n]);
      if constexpr (FUSED) {
        acc3[n]     = wmma_bf16(a0, b3f[n], acc3[n]);
        acc3[4 + n] = wmma_bf16(a1, b3f[n], acc3[4 + n]);
      }
    }
  };

#if USE_ASYNC_LDS
  auto stage_all = [&](int kt, int buf) {
    const int kk = kt * BK;
    stage_async_tile(Ab, kk, K, As[buf], c0r, c1r, c0c);
    stage_async_tile(B1b, kk, K, Bs1[buf], c0r, c1r, c0c);
    if constexpr (FUSED) stage_async_tile(B3b, kk, K, Bs3[buf], c0r, c1r, c0c);
  };

  // Prologue: two tiles in flight.
  stage_all(0, 0);
  stage_all(1, 1);
  __builtin_amdgcn_s_wait_asynccnt(FUSED ? 6 : 4);   // tile 0 complete
  __syncthreads();

  for (int t = 0; t < NT - 2; ++t) {
    const int cur = t % 3;
    stage_all(t + 2, (t + 2) % 3);    // keep two tiles of copies outstanding
    SBAR();                            // issue async staging first
    compute(cur);
    __builtin_amdgcn_s_wait_asynccnt(FUSED ? 6 : 4); // tile t+1 complete
    __syncthreads();
  }
  compute((NT - 2) % 3);
  __builtin_amdgcn_s_wait_asynccnt(0);               // tile NT-1 complete
  __syncthreads();
  compute((NT - 1) % 3);
#else
  // Fallback: 2-buffer manual staging.
  {
    uint4 qa0, qa1, qb0, qb1;
    ld2(Ab, 0, K, c0r, c1r, c0c, qa0, qa1);
    ld2(B1b, 0, K, c0r, c1r, c0c, qb0, qb1);
    st2(As[0], c0r, c1r, c0c, qa0, qa1);
    st2(Bs1[0], c0r, c1r, c0c, qb0, qb1);
    if constexpr (FUSED) {
      uint4 q30, q31;
      ld2(B3b, 0, K, c0r, c1r, c0c, q30, q31);
      st2(Bs3[0], c0r, c1r, c0c, q30, q31);
    }
  }
  __syncthreads();
  for (int t = 0; t < NT - 1; ++t) {
    const int cur = t & 1;
    const int kk  = (t + 1) * BK;
    uint4 qa0, qa1, qb0, qb1, q30 = {}, q31 = {};
    ld2(Ab, kk, K, c0r, c1r, c0c, qa0, qa1);
    ld2(B1b, kk, K, c0r, c1r, c0c, qb0, qb1);
    if constexpr (FUSED) ld2(B3b, kk, K, c0r, c1r, c0c, q30, q31);
    compute(cur);
    st2(As[1 - cur], c0r, c1r, c0c, qa0, qa1);
    st2(Bs1[1 - cur], c0r, c1r, c0c, qb0, qb1);
    if constexpr (FUSED) st2(Bs3[1 - cur], c0r, c1r, c0c, q30, q31);
    __syncthreads();
  }
  compute((NT - 1) & 1);
#endif

  // ---- epilogue: C/D layout VGPR v -> row v (+8 for hi half), lane -> column ----
  const int mrow0 = m0 + mw * 32 + hi * 8;
  const int col0  = n0 + nw * 64;
#pragma unroll
  for (int mi = 0; mi < 2; ++mi) {
#pragma unroll
    for (int n = 0; n < 4; ++n) {
      const int col = col0 + n * 16 + r;
#pragma unroll
      for (int v = 0; v < 8; ++v) {
        const int row = mrow0 + mi * 16 + v;
        if constexpr (FUSED) {
          const float g = acc1[mi * 4 + n][v];
          const float s = g / (1.0f + __expf(-g));   // silu
          Hout[(size_t)row * N + col] = f2bf(s * acc3[mi * 4 + n][v]);
        } else {
          Fout[(size_t)row * N + col] = acc1[mi * 4 + n][v];
        }
      }
    }
  }
}

extern "C" void kernel_launch(void* const* d_in, const int* in_sizes, int n_in,
                              void* d_out, int out_size, void* d_ws, size_t ws_size,
                              hipStream_t stream) {
  const float* x  = (const float*)d_in[0];
  const float* w1 = (const float*)d_in[1];
  const float* w2 = (const float*)d_in[2];
  const float* w3 = (const float*)d_in[3];
  const int*   ms = (const int*)d_in[4];
  float* out = (float*)d_out;

  // Workspace layout (240 MB total): xb, w1t, w3t, w2t (bf16, transposed), h (bf16).
  char* ws = (char*)d_ws;
  const size_t szX = (size_t)T_TOK * DIN * 2;
  const size_t szW = (size_t)NEXP * DIN * DHID * 2;
  u16* xb  = (u16*)(ws);
  u16* w1t = (u16*)(ws + szX);
  u16* w3t = (u16*)(ws + szX + szW);
  u16* w2t = (u16*)(ws + szX + 2 * szW);
  u16* hbf = (u16*)(ws + szX + 3 * szW);

  // 1) precision conversion passes
  {
    const int n = T_TOK * DIN;
    cvt_bf16_kernel<<<n / (256 * 4), 256, 0, stream>>>(x, xb, n);
  }
  {
    dim3 blk(32, 8);
    transpose_bf16_kernel<<<dim3(DHID / 32, DIN / 32, NEXP), blk, 0, stream>>>(w1, w1t, DIN, DHID);
    transpose_bf16_kernel<<<dim3(DHID / 32, DIN / 32, NEXP), blk, 0, stream>>>(w3, w3t, DIN, DHID);
    transpose_bf16_kernel<<<dim3(DIN / 32, DHID / 32, NEXP), blk, 0, stream>>>(w2, w2t, DHID, DIN);
  }

  // 2) fused grouped GEMM: h = silu(x@w1[e]) * (x@w3[e])
  grouped_gemm_kernel<true><<<dim3(DHID / 128, T_TOK / 128), 256, 0, stream>>>(
      xb, w1t, w3t, hbf, nullptr, ms, DIN, DHID);

  // 3) grouped GEMM: out = h @ w2[e]  (fp32 output)
  grouped_gemm_kernel<false><<<dim3(DIN / 128, T_TOK / 128), 256, 0, stream>>>(
      hbf, w2t, nullptr, nullptr, out, ms, DHID, DIN);
}
